// RNNEncoder_attn_70102456205386
// MI455X (gfx1250) — matrix operand order repaired and checked by reading, not compiled, not run on hardware
//
#include <hip/hip_runtime.h>
#include <hip/hip_bf16.h>

typedef __bf16 bf16;
typedef bf16 v16bf __attribute__((ext_vector_type(16)));
typedef bf16 v8bf  __attribute__((ext_vector_type(8)));
typedef float v8f  __attribute__((ext_vector_type(8)));

union V16 { v16bf v; v8bf h[2]; };

#define SEQT 512
#define NBL  192      // batch*layer
#define HIDN 768
#define HD   384
#define G4   1536
#define EPSF 1e-5f

__device__ __forceinline__ v8f wmma_bf16(v16bf a, v16bf b, v8f c) {
  return __builtin_amdgcn_wmma_f32_16x16x32_bf16(false, a, false, b, (short)0, c,
                                                 false, false);
}
__device__ __forceinline__ float sigf(float x) { return 1.0f / (1.0f + __expf(-x)); }

// ---------------------------------------------------------------- conversions
__global__ __launch_bounds__(256) void f32_to_bf16_k(const float* __restrict__ in,
                                                     bf16* __restrict__ out,
                                                     long long n) {
  long long i = ((long long)blockIdx.x * 256 + threadIdx.x) * 4;
  if (i + 3 < n) {
    float4 v = *(const float4*)(in + i);
    out[i + 0] = (bf16)v.x; out[i + 1] = (bf16)v.y;
    out[i + 2] = (bf16)v.z; out[i + 3] = (bf16)v.w;
  } else {
    for (; i < n; ++i) out[i] = (bf16)in[i];
  }
}

// ------------------------------------------------- input-projection WMMA GEMM
// Z[r][n] = sum_k A(row r)[k] * W[n][k] ; r -> (bl,t) via r%192,r/192 (or t_fixed)
#define BM 128
#define BN 128
#define BK 32
__global__ __launch_bounds__(256) void gemm_xW(const bf16* __restrict__ Ax,
                                               const bf16* __restrict__ Bw,
                                               float* __restrict__ Z,
                                               int M, int t_fixed) {
  __shared__ bf16 As[BM][BK];
  __shared__ bf16 Bs[BN][BK];
  const int nblk = blockIdx.x, mblk = blockIdx.y;
  const int tid = threadIdx.x;
  const int wave = tid >> 5, lane = tid & 31;
  const int wm = wave & 3, wn = wave >> 2;     // 4x2 wave grid -> 128x128 tile
  const int la = lane & 15;
  const int kb8 = (lane >> 4) * 8, kb16 = (lane >> 4) * 16;

  v8f acc[2][4];
  for (int a = 0; a < 2; ++a)
    for (int b = 0; b < 4; ++b)
      for (int e = 0; e < 8; ++e) acc[a][b][e] = 0.0f;

  const int lrow = tid >> 1;             // 0..127
  const int lhalf = (tid & 1) * 16;      // k sub-half (16 bf16 = 32B)
  const int grow = mblk * BM + lrow;
  const bool arow_ok = grow < M;
  const bf16* arow = Ax;
  if (arow_ok) {
    int bl, t;
    if (t_fixed >= 0) { bl = grow; t = t_fixed; }
    else { bl = grow % NBL; t = grow / NBL; }
    arow = Ax + ((size_t)bl * SEQT + t) * HIDN;
  }
  const bf16* brow = Bw + (size_t)(nblk * BN + lrow) * HIDN;

  for (int k0 = 0; k0 < HIDN; k0 += BK) {
    if (arow_ok) {
      const uint4* s = (const uint4*)(arow + k0 + lhalf);
      uint4* d = (uint4*)&As[lrow][lhalf];
      d[0] = s[0]; d[1] = s[1];
    } else {
      uint4 z4 = {0u, 0u, 0u, 0u};
      uint4* d = (uint4*)&As[lrow][lhalf];
      d[0] = z4; d[1] = z4;
    }
    {
      const uint4* s = (const uint4*)(brow + k0 + lhalf);
      uint4* d = (uint4*)&Bs[lrow][lhalf];
      d[0] = s[0]; d[1] = s[1];
    }
    __syncthreads();

    V16 af[2], bf[4];
    for (int sm = 0; sm < 2; ++sm) {
      const bf16* p = &As[wm * 32 + sm * 16 + la][0];
      af[sm].h[0] = *(const v8bf*)(p + kb8);
      af[sm].h[1] = *(const v8bf*)(p + kb8 + 16);
    }
    for (int sn = 0; sn < 4; ++sn) {
      const bf16* p = &Bs[wn * 64 + sn * 16 + la][0];
      bf[sn].h[0] = *(const v8bf*)(p + kb8);
      bf[sn].h[1] = *(const v8bf*)(p + kb8 + 16);
    }
    for (int sm = 0; sm < 2; ++sm)
      for (int sn = 0; sn < 4; ++sn)
        acc[sm][sn] = wmma_bf16(af[sm].v, bf[sn].v, acc[sm][sn]);
    __syncthreads();
  }

  const int m8 = (lane >> 4) * 8;
  for (int sm = 0; sm < 2; ++sm)
    for (int sn = 0; sn < 4; ++sn)
      for (int e = 0; e < 8; ++e) {
        int gm = mblk * BM + wm * 32 + sm * 16 + m8 + e;
        int gn = nblk * BN + wn * 64 + sn * 16 + la;
        if (gm < M) Z[(size_t)gm * G4 + gn] = acc[sm][sn][e];
      }
}

// --------------------------------------------------------------- row LayerNorm
// in-place: z = ((z + bias) - mean)*rstd*g + be   over N per row
__global__ __launch_bounds__(256) void ln_rows(float* __restrict__ Z,
                                               const float* __restrict__ bias,
                                               const float* __restrict__ g,
                                               const float* __restrict__ be,
                                               int N) {
  __shared__ float scr[16];
  __shared__ float s_mean, s_rstd;
  const size_t row = blockIdx.x;
  float* zr = Z + row * (size_t)N;
  const int tid = threadIdx.x, lane = tid & 31, wv = tid >> 5;
  float s1 = 0.f, s2 = 0.f;
  for (int j = tid; j < N; j += 256) {
    float v = zr[j] + bias[j];
    s1 += v; s2 += v * v;
  }
  for (int o = 16; o > 0; o >>= 1) { s1 += __shfl_xor(s1, o); s2 += __shfl_xor(s2, o); }
  if (lane == 0) { scr[wv * 2] = s1; scr[wv * 2 + 1] = s2; }
  __syncthreads();
  if (tid == 0) {
    float a = 0.f, b = 0.f;
    for (int w = 0; w < 8; ++w) { a += scr[w * 2]; b += scr[w * 2 + 1]; }
    float m = a / N;
    s_mean = m;
    s_rstd = rsqrtf(b / N - m * m + EPSF);
  }
  __syncthreads();
  float m = s_mean, r = s_rstd;
  for (int j = tid; j < N; j += 256) {
    float v = zr[j] + bias[j];
    zr[j] = (v - m) * r * g[j] + be[j];
  }
}

// --------------------------------------------------- forward LSTM recurrence
// 12 WGs x 16 rows; per step: gates = xg[t] + LN(h@Whh^T + bhh); lstm update.
__global__ __launch_bounds__(256) void lstm_fwd(const float* __restrict__ xg,
                                                const bf16* __restrict__ Whhb,
                                                const float* __restrict__ bhh0,
                                                const float* __restrict__ ghh,
                                                const float* __restrict__ behh,
                                                const float* __restrict__ gc,
                                                const float* __restrict__ bec,
                                                float* __restrict__ h_out) {
  extern __shared__ unsigned char smem[];
  bf16*  hbf  = (bf16*)smem;                    // 16x384 bf16   (12288 B)
  float* cmem = (float*)(smem + 12288);         // 16x384 f32    (24576 B)
  float* gmem = (float*)(smem + 36864);         // 16x1536 f32   (98304 B)
  float* stat = (float*)(smem + 135168);        // gm[16],gr[16],cm[16],cr[16]

  const int tid = threadIdx.x;
  const int wave = tid >> 5, lane = tid & 31;
  const int la = lane & 15;
  const int kb8 = (lane >> 4) * 8, kb16 = (lane >> 4) * 16;
  const int m8 = (lane >> 4) * 8;
  const int mbase = blockIdx.x * 16;

  for (int j = tid; j < 16 * HD; j += 256) { hbf[j] = (bf16)0.0f; cmem[j] = 0.0f; }
  __syncthreads();

  for (int t = 0; t < SEQT; ++t) {
    // ---- GEMM: gmem[m][n] = (h @ Whh^T)[m][n] + bhh[n]
    V16 afr[12];
    for (int ks = 0; ks < 12; ++ks) {
      const bf16* ap = hbf + la * HD + ks * 32;
      afr[ks].h[0] = *(const v8bf*)(ap + kb8);
      afr[ks].h[1] = *(const v8bf*)(ap + kb8 + 16);
    }
    for (int i = 0; i < 12; ++i) {
      const int nb = (wave * 12 + i) * 16;
      v8f acc; for (int e = 0; e < 8; ++e) acc[e] = 0.0f;
      const bf16* bbase = Whhb + (size_t)(nb + la) * HD;
      for (int ks = 0; ks < 12; ++ks) {
        V16 bfr;
        const bf16* bp = bbase + ks * 32 + kb8;
        bfr.h[0] = *(const v8bf*)bp;
        bfr.h[1] = *(const v8bf*)(bp + 16);
        acc = wmma_bf16(afr[ks].v, bfr.v, acc);
      }
      const int ncol = nb + la;
      const float bb = bhh0[ncol];
      for (int e = 0; e < 8; ++e) gmem[(m8 + e) * G4 + ncol] = acc[e] + bb;
    }
    __syncthreads();

    // ---- per-row LN stats over 1536 gates
    {
      const int r = tid >> 4, jj = tid & 15;
      float s1 = 0.f, s2 = 0.f;
      for (int q = 0; q < 96; ++q) {
        float v = gmem[r * G4 + jj + 16 * q];
        s1 += v; s2 += v * v;
      }
      for (int o = 8; o > 0; o >>= 1) { s1 += __shfl_xor(s1, o, 16); s2 += __shfl_xor(s2, o, 16); }
      if (jj == 0) {
        float m = s1 / G4;
        stat[r] = m;
        stat[16 + r] = rsqrtf(s2 / G4 - m * m + EPSF);
      }
    }
    __syncthreads();

    // ---- gates -> activations -> c update ; stash sigmoid(o) in o-slot
    {
      const size_t xb = ((size_t)t * NBL + mbase) * G4;
      for (int w = tid; w < 16 * HD; w += 256) {
        const int r = w / HD, j = w - r * HD;
        const float m = stat[r], rs = stat[16 + r];
        const float* gr = gmem + r * G4;
        const float* xr = xg + xb + (size_t)r * G4;
        float gi = (gr[j]            - m) * rs * ghh[j]            + behh[j]            + xr[j];
        float gf = (gr[HD + j]       - m) * rs * ghh[HD + j]       + behh[HD + j]       + xr[HD + j];
        float gg = (gr[2 * HD + j]   - m) * rs * ghh[2 * HD + j]   + behh[2 * HD + j]   + xr[2 * HD + j];
        float go = (gr[3 * HD + j]   - m) * rs * ghh[3 * HD + j]   + behh[3 * HD + j]   + xr[3 * HD + j];
        float c = sigf(gf) * cmem[r * HD + j] + sigf(gi) * tanhf(gg);
        cmem[r * HD + j] = c;
        gmem[r * G4 + 3 * HD + j] = sigf(go);
      }
    }
    __syncthreads();

    // ---- per-row LN stats over c (384)
    {
      const int r = tid >> 4, jj = tid & 15;
      float s1 = 0.f, s2 = 0.f;
      for (int q = 0; q < 24; ++q) {
        float v = cmem[r * HD + jj + 16 * q];
        s1 += v; s2 += v * v;
      }
      for (int o = 8; o > 0; o >>= 1) { s1 += __shfl_xor(s1, o, 16); s2 += __shfl_xor(s2, o, 16); }
      if (jj == 0) {
        float m = s1 / HD;
        stat[32 + r] = m;
        stat[48 + r] = rsqrtf(s2 / HD - m * m + EPSF);
      }
    }
    __syncthreads();

    // ---- h = sig(o) * tanh(LN(c)) ; write back as bf16 A operand
    for (int w = tid; w < 16 * HD; w += 256) {
      const int r = w / HD, j = w - r * HD;
      const float m = stat[32 + r], rs = stat[48 + r];
      float h = gmem[r * G4 + 3 * HD + j] *
                tanhf((cmem[r * HD + j] - m) * rs * gc[j] + bec[j]);
      hbf[r * HD + j] = (bf16)h;
      if (t == SEQT - 1) h_out[(size_t)(mbase + r) * HD + j] = h;
    }
    __syncthreads();
  }
}

// --------------------------------------- single backward step (h0 = c0 = 0)
__global__ __launch_bounds__(256) void lstm_bwd_last(const float* __restrict__ xgb,
                                                     const float* __restrict__ bhh1,
                                                     const float* __restrict__ ghh1,
                                                     const float* __restrict__ behh1,
                                                     const float* __restrict__ gc1,
                                                     const float* __restrict__ bec1,
                                                     float* __restrict__ h_out) {
  __shared__ float cbuf[HD], obuf[HD], scr[16];
  __shared__ float s_m, s_r;
  const int row = blockIdx.x, tid = threadIdx.x, lane = tid & 31, wv = tid >> 5;

  // LN stats of constant pre-activation vector bhh1 (h=0 -> gates_raw = bhh)
  float s1 = 0.f, s2 = 0.f;
  for (int j = tid; j < G4; j += 256) { float v = bhh1[j]; s1 += v; s2 += v * v; }
  for (int o = 16; o > 0; o >>= 1) { s1 += __shfl_xor(s1, o); s2 += __shfl_xor(s2, o); }
  if (lane == 0) { scr[wv * 2] = s1; scr[wv * 2 + 1] = s2; }
  __syncthreads();
  if (tid == 0) {
    float a = 0.f, b = 0.f;
    for (int w = 0; w < 8; ++w) { a += scr[w * 2]; b += scr[w * 2 + 1]; }
    float m = a / G4;
    s_m = m; s_r = rsqrtf(b / G4 - m * m + EPSF);
  }
  __syncthreads();
  const float mb = s_m, rb = s_r;
  const float* xr = xgb + (size_t)row * G4;
  for (int j = tid; j < HD; j += 256) {
    float gi = xr[j]          + (bhh1[j]          - mb) * rb * ghh1[j]          + behh1[j];
    float gg = xr[2 * HD + j] + (bhh1[2 * HD + j] - mb) * rb * ghh1[2 * HD + j] + behh1[2 * HD + j];
    float go = xr[3 * HD + j] + (bhh1[3 * HD + j] - mb) * rb * ghh1[3 * HD + j] + behh1[3 * HD + j];
    cbuf[j] = sigf(gi) * tanhf(gg);    // f * c0 = 0
    obuf[j] = sigf(go);
  }
  __syncthreads();
  s1 = 0.f; s2 = 0.f;
  for (int j = tid; j < HD; j += 256) { float v = cbuf[j]; s1 += v; s2 += v * v; }
  for (int o = 16; o > 0; o >>= 1) { s1 += __shfl_xor(s1, o); s2 += __shfl_xor(s2, o); }
  if (lane == 0) { scr[wv * 2] = s1; scr[wv * 2 + 1] = s2; }
  __syncthreads();
  if (tid == 0) {
    float a = 0.f, b = 0.f;
    for (int w = 0; w < 8; ++w) { a += scr[w * 2]; b += scr[w * 2 + 1]; }
    float m = a / HD;
    s_m = m; s_r = rsqrtf(b / HD - m * m + EPSF);
  }
  __syncthreads();
  const float mc = s_m, rc = s_r;
  for (int j = tid; j < HD; j += 256)
    h_out[(size_t)row * HD + j] = obuf[j] * tanhf((cbuf[j] - mc) * rc * gc1[j] + bec1[j]);
}

// ----------------------------------------------------- scoring + softmax(6)
__global__ __launch_bounds__(256) void score_softmax(const float* __restrict__ hf,
                                                     const float* __restrict__ hb,
                                                     const float* __restrict__ x,
                                                     const float* __restrict__ wo_w,
                                                     const float* __restrict__ wo_b,
                                                     float* __restrict__ scores) {
  __shared__ float sc[NBL];
  const int tid = threadIdx.x;
  if (tid < NBL) {
    const float* xl = x + ((size_t)tid * SEQT + (SEQT - 1)) * HIDN;
    float s = wo_b[0];
    for (int k = 0; k < HD; ++k) s += (hf[(size_t)tid * HD + k] + xl[k]) * wo_w[k];
    for (int k = 0; k < HD; ++k) s += (hb[(size_t)tid * HD + k] + xl[HD + k]) * wo_w[HD + k];
    sc[tid] = fmaxf(s, 0.0f);
  }
  __syncthreads();
  if (tid < 32) {
    float v[6], mx = -1e30f;
    for (int l = 0; l < 6; ++l) { v[l] = sc[tid * 6 + l]; mx = fmaxf(mx, v[l]); }
    float sum = 0.f;
    for (int l = 0; l < 6; ++l) { v[l] = __expf(v[l] - mx); sum += v[l]; }
    float inv = 1.0f / sum;
    for (int l = 0; l < 6; ++l) scores[tid * 6 + l] = v[l] * inv;
  }
}

// -------------------------------------------- out[b][s][h] = sum_l sc * x
__global__ __launch_bounds__(256) void weighted_sum(const float* __restrict__ x,
                                                    const float* __restrict__ scores,
                                                    float* __restrict__ out) {
  const long long total4 = (long long)32 * SEQT * (HIDN / 4);
  long long idx = (long long)blockIdx.x * 256 + threadIdx.x;
  if (idx >= total4) return;
  const int h4 = (int)(idx % (HIDN / 4));
  const int s  = (int)((idx / (HIDN / 4)) % SEQT);
  const int b  = (int)(idx / ((long long)(HIDN / 4) * SEQT));
  float4 acc = {0.f, 0.f, 0.f, 0.f};
  for (int l = 0; l < 6; ++l) {
    float w = scores[b * 6 + l];
    const float4* xp = (const float4*)(x + (((size_t)(b * 6 + l)) * SEQT + s) * HIDN) + h4;
    float4 v = *xp;
    acc.x += w * v.x; acc.y += w * v.y; acc.z += w * v.z; acc.w += w * v.w;
  }
  ((float4*)out)[idx] = acc;
}

// ============================================================= host launcher
extern "C" void kernel_launch(void* const* d_in, const int* in_sizes, int n_in,
                              void* d_out, int out_size, void* d_ws, size_t ws_size,
                              hipStream_t stream) {
  const float* x     = (const float*)d_in[0];
  const float* Wih   = (const float*)d_in[2];
  const float* Whh   = (const float*)d_in[3];
  const float* bih   = (const float*)d_in[4];
  const float* bhh   = (const float*)d_in[5];
  const float* g_ih  = (const float*)d_in[6];
  const float* be_ih = (const float*)d_in[7];
  const float* g_hh  = (const float*)d_in[8];
  const float* be_hh = (const float*)d_in[9];
  const float* g_c   = (const float*)d_in[10];
  const float* be_c  = (const float*)d_in[11];
  const float* wo_w  = (const float*)d_in[12];
  const float* wo_b  = (const float*)d_in[13];
  float* out = (float*)d_out;

  // workspace carve-up (256B aligned)
  size_t off = 0;
  auto carve = [&](size_t bytes) {
    void* p = (char*)d_ws + off;
    off += (bytes + 255) & ~(size_t)255;
    return p;
  };
  const long long NX   = (long long)32 * 6 * SEQT * HIDN;   // 75,497,472
  const long long NWIH = (long long)2 * G4 * HIDN;          //  2,359,296
  const long long NWHH = (long long)G4 * HD;                //    589,824 (dir0)
  bf16*  xb    = (bf16*)carve(NX * 2);
  bf16*  wihb  = (bf16*)carve(NWIH * 2);
  bf16*  whhb  = (bf16*)carve(NWHH * 2);
  float* xg    = (float*)carve((size_t)SEQT * NBL * G4 * 4);  // fwd proj (LN in place)
  float* xgb   = (float*)carve((size_t)NBL * G4 * 4);         // bwd proj @ t=511
  float* hf    = (float*)carve((size_t)NBL * HD * 4);
  float* hb    = (float*)carve((size_t)NBL * HD * 4);
  float* sc    = (float*)carve((size_t)NBL * 4);

  // 1) fp32 -> bf16 conversions
  f32_to_bf16_k<<<(unsigned)((NX / 4 + 255) / 256), 256, 0, stream>>>(x, xb, NX);
  f32_to_bf16_k<<<(unsigned)((NWIH / 4 + 255) / 256), 256, 0, stream>>>(Wih, wihb, NWIH);
  f32_to_bf16_k<<<(unsigned)((NWHH / 4 + 255) / 256), 256, 0, stream>>>(Whh, whhb, NWHH);

  // 2) forward input projection GEMM + LN  (M = 512*192)
  {
    dim3 grid(G4 / BN, (SEQT * NBL) / BM);
    gemm_xW<<<grid, 256, 0, stream>>>(xb, wihb, xg, SEQT * NBL, -1);
    ln_rows<<<SEQT * NBL, 256, 0, stream>>>(xg, bih, g_ih, be_ih, G4);
  }

  // 3) backward input projection only at t = SEQ-1 (M = 192)
  {
    dim3 grid(G4 / BN, (NBL + BM - 1) / BM);
    gemm_xW<<<grid, 256, 0, stream>>>(xb, wihb + (size_t)G4 * HIDN, xgb, NBL, SEQT - 1);
    ln_rows<<<NBL, 256, 0, stream>>>(xgb, bih + G4, g_ih + G4, be_ih + G4, G4);
  }

  // 4) single backward step (dir 1)
  lstm_bwd_last<<<NBL, 256, 0, stream>>>(xgb, bhh + G4, g_hh + G4, be_hh + G4,
                                         g_c + HD, be_c + HD, hb);

  // 5) full forward recurrence (dir 0), 12 persistent WGs of 16 rows
  {
    const int SMEM = 135168 + 64 * 4;  // hbf + c + gates + stats
    hipFuncSetAttribute((const void*)lstm_fwd,
                        hipFuncAttributeMaxDynamicSharedMemorySize, SMEM);
    lstm_fwd<<<NBL / 16, 256, SMEM, stream>>>(xg, whhb, bhh, g_hh, be_hh,
                                              g_c, be_c, hf);
  }

  // 6) scores + softmax over layers
  score_softmax<<<1, 256, 0, stream>>>(hf, hb, x, wo_w, wo_b, sc);

  // 7) sent_vec = sum_l scores * x
  {
    const long long total4 = (long long)32 * SEQT * (HIDN / 4);
    weighted_sum<<<(unsigned)((total4 + 255) / 256), 256, 0, stream>>>(x, sc, out);
  }
}
